// SpatialAttentionLayer_63187558858867
// MI455X (gfx1250) — matrix operand order repaired and checked by reading, not compiled, not run on hardware
//
#include <hip/hip_runtime.h>
#include <hip/hip_bf16.h>
#include <stddef.h>

#define NU 100000
#define NI 50000
#define EDIM 32
#define NE_RATE 1000000
#define NE_TRUST 500000

typedef __attribute__((ext_vector_type(16))) _Float16 v16h;
typedef __attribute__((ext_vector_type(8)))  float    v8f;

static inline int ceil_div(int a, int b) { return (a + b - 1) / b; }

// ---------------- fill kernels (d_ws is poisoned; we must init) ----------------
__global__ __launch_bounds__(256) void fill_f32(float* __restrict__ p, float v, int n) {
  int i = blockIdx.x * blockDim.x + threadIdx.x;
  if (i < n) p[i] = v;
}
__global__ __launch_bounds__(256) void fill_u32(unsigned* __restrict__ p, unsigned v, int n) {
  int i = blockIdx.x * blockDim.x + threadIdx.x;
  if (i < n) p[i] = v;
}

// ---------------- WMMA projection: out[N x 32] = h[N x 32] @ W^T + bias ----------------
// One wave32 computes a 16-row x 32-col tile with two v_wmma_f32_16x16x32_f16.
// A layout (16x32 f16): lane = 16*hs + r holds row M=r; element t -> K = ((t<8)?t:t+8) + 8*hs
// B layout (32x16 f16): lane = 16*hs + r holds column N=r; element t -> K = ((t<8)?t:t+8) + 8*hs
// C/D layout (16x16 f32): lane = 16*hs + r, element i -> (M = i + 8*hs, N = r)
__global__ __launch_bounds__(256) void gat_proj_wmma(const float* __restrict__ h,
                                                     const float* __restrict__ W,
                                                     const float* __restrict__ bias,
                                                     float* __restrict__ out, int n) {
  int wave = (int)((blockIdx.x * blockDim.x + threadIdx.x) >> 5);
  int lane = threadIdx.x & 31;
  int row0 = wave * 16;
  if (row0 >= n) return;          // n is a multiple of 16 for all call sites
  int hs = lane >> 4;
  int r  = lane & 15;

  const float* hrow = h + (size_t)(row0 + r) * EDIM;
  v16h A;
#pragma unroll
  for (int t = 0; t < 16; ++t) {
    int k = ((t < 8) ? t : t + 8) + 8 * hs;
    A[t] = (_Float16)hrow[k];
  }

  int kb = 8 * hs;                // K offset of this lane half for B (same order as A)
  v16h B0, B1;
#pragma unroll
  for (int t = 0; t < 16; ++t) {
    int k = ((t < 8) ? t : t + 8) + kb;
    B0[t] = (_Float16)W[(size_t)r * EDIM + k];          // (W^T)[k][r] = W[r][k], cols 0..15
    B1[t] = (_Float16)W[(size_t)(r + 16) * EDIM + k];   // cols 16..31
  }

  float b0 = bias[r], b1 = bias[r + 16];
  v8f C0, C1;
#pragma unroll
  for (int i = 0; i < 8; ++i) { C0[i] = b0; C1[i] = b1; }

  C0 = __builtin_amdgcn_wmma_f32_16x16x32_f16(false, A, false, B0, (short)0, C0, false, false);
  C1 = __builtin_amdgcn_wmma_f32_16x16x32_f16(false, A, false, B1, (short)0, C1, false, false);

  float* orow = out + (size_t)row0 * EDIM;
#pragma unroll
  for (int i = 0; i < 8; ++i) {
    int m = i + 8 * hs;
    orow[(size_t)m * EDIM + r]      = C0[i];
    orow[(size_t)m * EDIM + r + 16] = C1[i];
  }
}

// ---------------- edge pass 1: logits + segment max (monotonic uint key) ----------------
__device__ __forceinline__ unsigned f32_key(float f) {
  unsigned b = __float_as_uint(f);
  return (b & 0x80000000u) ? ~b : (b | 0x80000000u);
}
__device__ __forceinline__ float key_f32(unsigned k) {
  unsigned b = (k & 0x80000000u) ? (k ^ 0x80000000u) : ~k;
  return __uint_as_float(b);
}

__global__ __launch_bounds__(256) void edge_logits(const int* __restrict__ src,
                                                   const int* __restrict__ dst,
                                                   const float* __restrict__ fs,
                                                   const float* __restrict__ fd,
                                                   const float* __restrict__ attn,
                                                   float* __restrict__ e,
                                                   unsigned* __restrict__ mkey, int ne) {
  int edge = (int)((blockIdx.x * blockDim.x + threadIdx.x) >> 5);
  int lane = threadIdx.x & 31;
  if (edge >= ne) return;
  int s = src[edge], d = dst[edge];
  float v = fs[(size_t)s * EDIM + lane] + fd[(size_t)d * EDIM + lane];
  v = (v > 0.f) ? v : 0.2f * v;               // leaky_relu(., 0.2)
  float p = v * attn[lane];
#pragma unroll
  for (int off = 16; off > 0; off >>= 1) p += __shfl_xor(p, off, 32);
  if (lane == 0) {
    e[edge] = p;
    atomicMax(&mkey[d], f32_key(p));
  }
}

// ---------------- edge pass 2: ex = exp(e - m[dst]); s[dst] += ex ----------------
__global__ __launch_bounds__(256) void edge_softmax_sum(const int* __restrict__ dst,
                                                        float* __restrict__ e,
                                                        const unsigned* __restrict__ mkey,
                                                        float* __restrict__ ssum, int ne) {
  int i = blockIdx.x * blockDim.x + threadIdx.x;
  if (i >= ne) return;
  int d = dst[i];
  float m = key_f32(mkey[d]);
  float ex = __expf(e[i] - m);
  e[i] = ex;
  atomicAdd(&ssum[d], ex);
}

// ---------------- edge pass 3: out[dst] += fs[src] * ex (un-normalized) ----------------
__global__ __launch_bounds__(256) void edge_aggregate(const int* __restrict__ src,
                                                      const int* __restrict__ dst,
                                                      const float* __restrict__ fs,
                                                      const float* __restrict__ e,
                                                      float* __restrict__ out, int ne) {
  int edge = (int)((blockIdx.x * blockDim.x + threadIdx.x) >> 5);
  int lane = threadIdx.x & 31;
  if (edge >= ne) return;
  int s = src[edge], d = dst[edge];
  float ex = e[edge];
  atomicAdd(&out[(size_t)d * EDIM + lane], fs[(size_t)s * EDIM + lane] * ex);
}

// ---------------- normalize by segment sum (0 for empty segments) ----------------
__global__ __launch_bounds__(256) void seg_normalize(float* __restrict__ out,
                                                     const float* __restrict__ ssum, int nd) {
  int i = blockIdx.x * blockDim.x + threadIdx.x;
  if (i >= nd * EDIM) return;
  float s = ssum[i >> 5];
  out[i] = (s > 0.f) ? out[i] / s : 0.f;
}

// ---------------- output head: y = [a|b] @ Wout^T + bout (WMMA, K=64 = 2x32) ----------------
__global__ __launch_bounds__(256) void head_wmma(const float* __restrict__ a,
                                                 const float* __restrict__ b,
                                                 const float* __restrict__ Wout,  // [32 x 64]
                                                 const float* __restrict__ bout,
                                                 float* __restrict__ y, int n) {
  int wave = (int)((blockIdx.x * blockDim.x + threadIdx.x) >> 5);
  int lane = threadIdx.x & 31;
  int row0 = wave * 16;
  if (row0 >= n) return;
  int hs = lane >> 4;
  int r  = lane & 15;

  const float* arow = a + (size_t)(row0 + r) * EDIM;
  const float* brow = b + (size_t)(row0 + r) * EDIM;
  v16h Aa, Ab;
#pragma unroll
  for (int t = 0; t < 16; ++t) {
    int k = ((t < 8) ? t : t + 8) + 8 * hs;
    Aa[t] = (_Float16)arow[k];
    Ab[t] = (_Float16)brow[k];
  }

  int kb = 8 * hs;
  v16h B00, B01, B10, B11;   // B{kblock}{coltile}
#pragma unroll
  for (int t = 0; t < 16; ++t) {
    int k = ((t < 8) ? t : t + 8) + kb;
    B00[t] = (_Float16)Wout[(size_t)r * 64 + k];
    B01[t] = (_Float16)Wout[(size_t)(r + 16) * 64 + k];
    B10[t] = (_Float16)Wout[(size_t)r * 64 + 32 + k];
    B11[t] = (_Float16)Wout[(size_t)(r + 16) * 64 + 32 + k];
  }

  float b0 = bout[r], b1 = bout[r + 16];
  v8f C0, C1;
#pragma unroll
  for (int i = 0; i < 8; ++i) { C0[i] = b0; C1[i] = b1; }

  C0 = __builtin_amdgcn_wmma_f32_16x16x32_f16(false, Aa, false, B00, (short)0, C0, false, false);
  C0 = __builtin_amdgcn_wmma_f32_16x16x32_f16(false, Ab, false, B10, (short)0, C0, false, false);
  C1 = __builtin_amdgcn_wmma_f32_16x16x32_f16(false, Aa, false, B01, (short)0, C1, false, false);
  C1 = __builtin_amdgcn_wmma_f32_16x16x32_f16(false, Ab, false, B11, (short)0, C1, false, false);

  float* yrow = y + (size_t)row0 * EDIM;
#pragma unroll
  for (int i = 0; i < 8; ++i) {
    int m = i + 8 * hs;
    yrow[(size_t)m * EDIM + r]      = C0[i];
    yrow[(size_t)m * EDIM + r + 16] = C1[i];
  }
}

// ---------------- batchnorm stats (per-column sum / sumsq over NU rows) ----------------
__global__ __launch_bounds__(256) void bn_stats(const float* __restrict__ y,
                                                float* __restrict__ sums,
                                                float* __restrict__ sqs, int n) {
  int col  = threadIdx.x & 31;
  int rlan = threadIdx.x >> 5;                 // 8 row-groups per block
  float s = 0.f, q = 0.f;
  for (int row = blockIdx.x * 8 + rlan; row < n; row += gridDim.x * 8) {
    float v = y[(size_t)row * EDIM + col];
    s += v; q += v * v;
  }
  __shared__ float ls[256], lq[256];
  ls[threadIdx.x] = s; lq[threadIdx.x] = q;
  __syncthreads();
  if (threadIdx.x < 32) {
#pragma unroll
    for (int g = 1; g < 8; ++g) { s += ls[col + g * 32]; q += lq[col + g * 32]; }
    atomicAdd(&sums[col], s);
    atomicAdd(&sqs[col], q);
  }
}

// ---------------- batchnorm apply + LeakyReLU(0.01), in place ----------------
__global__ __launch_bounds__(256) void bn_apply(float* __restrict__ y,
                                                const float* __restrict__ sums,
                                                const float* __restrict__ sqs,
                                                const float* __restrict__ gamma,
                                                const float* __restrict__ beta, int n) {
  int i = blockIdx.x * blockDim.x + threadIdx.x;
  if (i >= n * EDIM) return;
  int col = i & 31;
  float invn = 1.0f / (float)n;
  float mean = sums[col] * invn;
  float var  = sqs[col] * invn - mean * mean;
  float v = (y[i] - mean) * rsqrtf(var + 1e-5f) * gamma[col] + beta[col];
  y[i] = (v > 0.f) ? v : 0.01f * v;
}

// ---------------- one GATv2 relation ----------------
static void run_gat(hipStream_t stream,
                    const float* hsrc, int ns, const float* hdst, int nd,
                    const int* srcIdx, const int* dstIdx, int ne,
                    const float* Wsrc, const float* bsrc,
                    const float* Wdst, const float* bdst, const float* attn,
                    float* fs, float* fd, float* e, unsigned* mkey, float* ssum,
                    float* out) {
  gat_proj_wmma<<<ceil_div(ns / 16, 8), 256, 0, stream>>>(hsrc, Wsrc, bsrc, fs, ns);
  gat_proj_wmma<<<ceil_div(nd / 16, 8), 256, 0, stream>>>(hdst, Wdst, bdst, fd, nd);
  fill_u32<<<ceil_div(nd, 256), 256, 0, stream>>>(mkey, 0u, nd);
  fill_f32<<<ceil_div(nd, 256), 256, 0, stream>>>(ssum, 0.f, nd);
  fill_f32<<<ceil_div(nd * EDIM, 256), 256, 0, stream>>>(out, 0.f, nd * EDIM);
  edge_logits<<<ceil_div(ne, 8), 256, 0, stream>>>(srcIdx, dstIdx, fs, fd, attn, e, mkey, ne);
  edge_softmax_sum<<<ceil_div(ne, 256), 256, 0, stream>>>(dstIdx, e, mkey, ssum, ne);
  edge_aggregate<<<ceil_div(ne, 8), 256, 0, stream>>>(srcIdx, dstIdx, fs, e, out, ne);
  seg_normalize<<<ceil_div(nd * EDIM, 256), 256, 0, stream>>>(out, ssum, nd);
}

extern "C" void kernel_launch(void* const* d_in, const int* in_sizes, int n_in,
                              void* d_out, int out_size, void* d_ws, size_t ws_size,
                              hipStream_t stream) {
  (void)in_sizes; (void)n_in; (void)out_size; (void)ws_size;

  const float* user_embed = (const float*)d_in[0];
  const float* item_embed = (const float*)d_in[1];
  const int*   rate_src   = (const int*)d_in[2];
  const int*   rate_dst   = (const int*)d_in[3];
  const int*   trust_src  = (const int*)d_in[4];
  const int*   trust_dst  = (const int*)d_in[5];

  // relation params: 5 per relation starting at index 6
  const float* P[20];
  for (int i = 0; i < 20; ++i) P[i] = (const float*)d_in[6 + i];
  // P[0..4]  = l1_rate  {Wsrc,bsrc,Wdst,bdst,attn}
  // P[5..9]  = l1_rby
  // P[10..14]= l2_rby
  // P[15..19]= l2_trust
  const float* Wout  = (const float*)d_in[26];
  const float* bout  = (const float*)d_in[27];
  const float* gamma = (const float*)d_in[28];
  const float* beta  = (const float*)d_in[29];
  float* y = (float*)d_out;

  // workspace carve-up (floats)
  float* ws = (float*)d_ws;
  size_t o = 0;
  float*    h1    = ws + o; o += (size_t)NI * EDIM;
  float*    h2    = ws + o; o += (size_t)NU * EDIM;
  float*    abuf  = ws + o; o += (size_t)NU * EDIM;
  float*    bbuf  = ws + o; o += (size_t)NU * EDIM;
  float*    fs    = ws + o; o += (size_t)NU * EDIM;
  float*    fd    = ws + o; o += (size_t)NU * EDIM;
  float*    e     = ws + o; o += (size_t)NE_RATE;
  unsigned* mkey  = (unsigned*)(ws + o); o += (size_t)NU;
  float*    ssum  = ws + o; o += (size_t)NU;
  float*    stats = ws + o; o += 64;

  // layer 1: user -> item ('rate')
  run_gat(stream, user_embed, NU, item_embed, NI, rate_src, rate_dst, NE_RATE,
          P[0], P[1], P[2], P[3], P[4], fs, fd, e, mkey, ssum, h1);
  // layer 1: item -> user ('rated-by')
  run_gat(stream, item_embed, NI, user_embed, NU, rate_dst, rate_src, NE_RATE,
          P[5], P[6], P[7], P[8], P[9], fs, fd, e, mkey, ssum, h2);
  // layer 2: item-influence (h1 via 'rated-by')
  run_gat(stream, h1, NI, user_embed, NU, rate_dst, rate_src, NE_RATE,
          P[10], P[11], P[12], P[13], P[14], fs, fd, e, mkey, ssum, abuf);
  // layer 2: social (h2 via 'trust')
  run_gat(stream, h2, NU, user_embed, NU, trust_src, trust_dst, NE_TRUST,
          P[15], P[16], P[17], P[18], P[19], fs, fd, e, mkey, ssum, bbuf);

  // output head: Linear(2E,E) via WMMA -> BatchNorm (training stats) -> LeakyReLU(0.01)
  head_wmma<<<ceil_div(NU / 16, 8), 256, 0, stream>>>(abuf, bbuf, Wout, bout, y, NU);
  fill_f32<<<1, 64, 0, stream>>>(stats, 0.f, 64);
  bn_stats<<<512, 256, 0, stream>>>(y, stats, stats + 32, NU);
  bn_apply<<<ceil_div(NU * EDIM, 256), 256, 0, stream>>>(y, stats, stats + 32, gamma, beta, NU);
}
